// Indexing_Blend_Kron3_MLP_77223511982254
// MI455X (gfx1250) — matrix operand order repaired and checked by reading, not compiled, not run on hardware
//
#include <hip/hip_runtime.h>
#include <hip/hip_bf16.h>
#include <math.h>

// ---------------------------------------------------------------------------
// Problem constants (from reference)
// ---------------------------------------------------------------------------
#define NGRID   64          // grid resolution per axis
#define W0C     256         // width0 (table feature dim)
#define WIDTHC  256         // hidden width
#define OUTC    3
#define MQ      131072      // number of query points
#define KNB     8           // blend neighbors
#define NROWS   (NGRID*NGRID*NGRID)   // 262144 table rows

// workspace layout (bytes)
#define TABLE_BYTES ((size_t)NROWS * W0C * 2)          // f16 table = 128 MB (fits L2)
#define W1T_HALFS   (WIDTHC * W0C)                      // 256x256 f16, N-major
#define W2T_HALFS   (16 * WIDTHC)                       // 16x256 f16 (cols 3..15 zero)

typedef __attribute__((ext_vector_type(16))) _Float16 v16h;
typedef __attribute__((ext_vector_type(8)))  _Float16 v8h;
typedef __attribute__((ext_vector_type(8)))  float    v8f;

#define QB      128         // queries per block in fused kernel
#define HSTRIDE 264         // padded LDS row stride (halfs): 132 dwords -> bank step 4

// ---------------------------------------------------------------------------
// Kernel 0: convert + transpose weights to f16 (W1T[n][k], W2T[n][k] padded)
// ---------------------------------------------------------------------------
__global__ __launch_bounds__(256)
void prep_weights(const float* __restrict__ W1, const float* __restrict__ W2,
                  _Float16* __restrict__ W1T, _Float16* __restrict__ W2T)
{
    const int b = blockIdx.x;
    const int t = threadIdx.x;
    if (b < WIDTHC) {
        // W1: (256 in, 256 out) row-major -> W1T[n*256 + k] = W1[k*256 + n]
        W1T[b * W0C + t] = (_Float16)W1[t * WIDTHC + b];
    } else {
        const int n = b - WIDTHC;       // 0..15
        W2T[n * WIDTHC + t] = (n < OUTC) ? (_Float16)W2[t * OUTC + n] : (_Float16)0.0f;
    }
}

// ---------------------------------------------------------------------------
// Kernel 1: build table (f16).  One block per (i,k), thread = w.
// table[(i*64+k)*64 + j][w] = sum_{a,b,c} x[i,c]*first[w,a,c,b]*y[j,b]*z[k,a]
// Stores are coalesced 512B rows across the 256 threads.
// ---------------------------------------------------------------------------
__global__ __launch_bounds__(256)
void build_table(const float* __restrict__ x, const float* __restrict__ y,
                 const float* __restrict__ z, const float* __restrict__ first,
                 _Float16* __restrict__ table)
{
    const int w   = threadIdx.x;         // 0..255
    const int blk = blockIdx.x;          // i*64 + k
    const int i   = blk >> 6;
    const int k   = blk & 63;

    float f[27];                         // first[w][a][c][b]
    #pragma unroll
    for (int q = 0; q < 27; ++q) f[q] = first[w * 27 + q];

    const float x0 = x[i*3+0], x1 = x[i*3+1], x2 = x[i*3+2];
    const float z0 = z[k*3+0], z1 = z[k*3+1], z2 = z[k*3+2];

    // s1[a][b] = sum_c first[w,a,c,b] * x[i,c]
    float s1[9];
    #pragma unroll
    for (int a = 0; a < 3; ++a)
        #pragma unroll
        for (int b = 0; b < 3; ++b)
            s1[a*3+b] = fmaf(f[a*9 + b], x0,
                        fmaf(f[a*9 + 3 + b], x1,
                             f[a*9 + 6 + b] * x2));

    const size_t nbase = (size_t)blk * 64;
    for (int j = 0; j < 64; ++j) {
        const float y0 = y[j*3+0], y1 = y[j*3+1], y2 = y[j*3+2];   // scalar loads (uniform)
        const float t0 = fmaf(s1[0], y0, fmaf(s1[1], y1, s1[2]*y2));
        const float t1 = fmaf(s1[3], y0, fmaf(s1[4], y1, s1[5]*y2));
        const float t2 = fmaf(s1[6], y0, fmaf(s1[7], y1, s1[8]*y2));
        const float val = fmaf(z0, t0, fmaf(z1, t1, z2 * t2));
        table[(nbase + j) * W0C + w] = (_Float16)val;
    }
}

// ---------------------------------------------------------------------------
// helpers: WMMA fragment loads per CDNA5 16-bit layouts (CDNA5 ISA)
// A (16x32 f16): lane row = lnm; K base = (lane<16 ? 0 : 8); halfs {K..K+7, K+16..K+23}
// B (32x16 f16): lane col = lnm; K base = (lane<16 ? 0 : 8); halfs {K..K+7, K+16..K+23}
// ---------------------------------------------------------------------------
__device__ __forceinline__ v16h load_a_frag(const _Float16* rowptr, int k0, int hi) {
    const _Float16* p = rowptr + k0 + hi * 8;
    v8h lo = *(const v8h*)(p);
    v8h hi8 = *(const v8h*)(p + 16);
    return __builtin_shufflevector(lo, hi8, 0,1,2,3,4,5,6,7,8,9,10,11,12,13,14,15);
}
__device__ __forceinline__ v16h load_b_frag(const _Float16* colptr, int k0, int hi) {
    const _Float16* p = colptr + k0 + hi * 8;
    v8h lo = *(const v8h*)(p);
    v8h hi8 = *(const v8h*)(p + 16);
    return __builtin_shufflevector(lo, hi8, 0,1,2,3,4,5,6,7,8,9,10,11,12,13,14,15);
}

// ---------------------------------------------------------------------------
// Kernel 2 (fused): gather-blend -> ReLU -> GEMM1(WMMA)+ReLU -> GEMM2(WMMA)
//                   -> sigmoid -> out.   128 queries / block, 8 waves.
// ---------------------------------------------------------------------------
__global__ __launch_bounds__(256)
void fused_blend_mlp(const int*   __restrict__ Bidx, const float* __restrict__ Bw,
                     const _Float16* __restrict__ table,
                     const _Float16* __restrict__ W1T, const _Float16* __restrict__ W2T,
                     const float* __restrict__ b1,   const float* __restrict__ b2,
                     float* __restrict__ out)
{
    __shared__ __align__(16) _Float16 H [QB * HSTRIDE];
    __shared__ __align__(16) _Float16 H2[QB * HSTRIDE];

    const int t  = threadIdx.x;
    const int m0 = blockIdx.x * QB;

    // ---------------- phase A: weighted gather from f16 table (L2-resident) ----
    {
        const int q    = t >> 1;             // query within block
        const int half = (t & 1) * 128;      // feature half handled by this thread
        const int m    = m0 + q;
        int   idx[KNB];
        float wt [KNB];
        #pragma unroll
        for (int k = 0; k < KNB; ++k) {
            idx[k] = Bidx[m * KNB + k];
            wt [k] = Bw  [m * KNB + k];
        }
        #pragma unroll
        for (int seg = 0; seg < 16; ++seg) {
            const int wb = half + seg * 8;
            float acc[8] = {0,0,0,0,0,0,0,0};
            #pragma unroll
            for (int k = 0; k < KNB; ++k) {
                v8h v = *(const v8h*)(table + (size_t)idx[k] * W0C + wb);
                #pragma unroll
                for (int j = 0; j < 8; ++j)
                    acc[j] = fmaf(wt[k], (float)v[j], acc[j]);
            }
            v8h o;
            #pragma unroll
            for (int j = 0; j < 8; ++j)
                o[j] = (_Float16)fmaxf(acc[j], 0.0f);          // ReLU before MLP
            *(v8h*)(H + q * HSTRIDE + wb) = o;
        }
    }
    __syncthreads();

    const int wave = t >> 5;
    const int lane = t & 31;
    const int lnm  = lane & 15;
    const int hi   = lane >> 4;          // 0 | 1 (half-wave)
    const int mrow = wave * 16;          // this wave's 16-row strip

    // ---------------- phase B: H(128x256) @ W1 + b1, ReLU -> H2 ---------------
    for (int n0 = 0; n0 < WIDTHC; n0 += 16) {
        const float bb = b1[n0 + lnm];               // matches f32 C/D layout (N=lane%16)
        v8f c = {bb, bb, bb, bb, bb, bb, bb, bb};
        #pragma unroll
        for (int k0 = 0; k0 < W0C; k0 += 32) {
            v16h a = load_a_frag(H + (mrow + lnm) * HSTRIDE, k0, hi);
            v16h b = load_b_frag(W1T + (n0 + lnm) * W0C, k0, hi);
            c = __builtin_amdgcn_wmma_f32_16x16x32_f16(false, a, false, b,
                                                       (short)0, c, false, false);
        }
        #pragma unroll
        for (int r = 0; r < 8; ++r) {
            const float v = fmaxf(c[r], 0.0f);       // row = mrow + r + hi*8, col = n0+lnm
            H2[(mrow + r + hi * 8) * HSTRIDE + n0 + lnm] = (_Float16)v;
        }
    }
    __syncthreads();

    // ---------------- phase C: H2 @ W2 (padded to 16 cols) + b2, sigmoid ------
    {
        const float bb = (lnm < OUTC) ? b2[lnm] : 0.0f;
        v8f c = {bb, bb, bb, bb, bb, bb, bb, bb};
        #pragma unroll
        for (int k0 = 0; k0 < WIDTHC; k0 += 32) {
            v16h a = load_a_frag(H2 + (mrow + lnm) * HSTRIDE, k0, hi);
            v16h b = load_b_frag(W2T + lnm * WIDTHC, k0, hi);
            c = __builtin_amdgcn_wmma_f32_16x16x32_f16(false, a, false, b,
                                                       (short)0, c, false, false);
        }
        if (lnm < OUTC) {
            #pragma unroll
            for (int r = 0; r < 8; ++r) {
                const float s = 1.0f / (1.0f + __expf(-c[r]));
                out[(size_t)(m0 + mrow + r + hi * 8) * OUTC + lnm] = s;
            }
        }
    }
}

// ---------------------------------------------------------------------------
// launch
// ---------------------------------------------------------------------------
extern "C" void kernel_launch(void* const* d_in, const int* in_sizes, int n_in,
                              void* d_out, int out_size, void* d_ws, size_t ws_size,
                              hipStream_t stream) {
    const int*   Bidx  = (const int*)  d_in[0];
    const float* Bw    = (const float*)d_in[1];
    const float* x     = (const float*)d_in[2];
    const float* y     = (const float*)d_in[3];
    const float* z     = (const float*)d_in[4];
    const float* first = (const float*)d_in[5];
    const float* W1    = (const float*)d_in[6];
    const float* b1    = (const float*)d_in[7];
    const float* W2    = (const float*)d_in[8];
    const float* b2    = (const float*)d_in[9];
    float* out = (float*)d_out;

    char* ws = (char*)d_ws;
    _Float16* table = (_Float16*)ws;                       // 128 MB f16 table
    _Float16* W1T   = (_Float16*)(ws + TABLE_BYTES);       // 256x256 f16 (N-major)
    _Float16* W2T   = W1T + W1T_HALFS;                     // 16x256 f16 (zero-padded)

    prep_weights<<<WIDTHC + 16, 256, 0, stream>>>(W1, W2, W1T, W2T);
    build_table <<<NGRID * NGRID, 256, 0, stream>>>(x, y, z, first, table);
    fused_blend_mlp<<<MQ / QB, 256, 0, stream>>>(Bidx, Bw, table, W1T, W2T, b1, b2, out);
}
